// GATv2_23398981828938
// MI455X (gfx1250) — matrix operand order repaired and checked by reading, not compiled, not run on hardware
//
#include <hip/hip_runtime.h>

#define N_NODES 50000
#define E_EDGES 800000
#define K_DIM   128     // fan-in of every layer
#define HIDC    64

typedef __attribute__((ext_vector_type(16))) _Float16 v16h;
typedef __attribute__((ext_vector_type(8)))  _Float16 v8h;
typedef __attribute__((ext_vector_type(8)))  float    v8f;

// ---------------------------------------------------------------------------
// float atomic max via CAS (segment_max)
// ---------------------------------------------------------------------------
__device__ __forceinline__ void atomicMaxF(float* addr, float val) {
    unsigned int* ua  = (unsigned int*)addr;
    unsigned int  old = *ua;
    while (__uint_as_float(old) < val) {
        unsigned int assumed = old;
        old = atomicCAS(ua, assumed, __float_as_uint(val));
        if (old == assumed) break;
    }
}

// ---------------------------------------------------------------------------
// WMMA GEMM, one block per 16-row m-tile, computing BOTH xl = x@Wl and
// xr = x@Wr for all HC output columns.
//  * x tile (16 x 128 fp32) pulled global->LDS with CDNA5 async-DMA
//    (global_load_async_to_lds_b128, ASYNCcnt), converted once to f16.
//  * Wl/Wr staged transposed+f16 in LDS, rows padded to 136 halfs (272 B,
//    16B-aligned, bank-spread) so fragments are pure ds_load_b128.
//  * 4 waves: wave = {side(Wl/Wr) x n-half}; each wave keeps TPW = HC/32
//    f32 accumulators and reuses one A fragment per k-step.
//  Fragment layouts per CDNA5 ISA 7.12.2 (16x16x32 f16):
//    A: lanes 0-15 row M=lane, K {0..7,16..23}; lanes 16-31 K {8..15,24..31}
//    B: lanes 0-15 col N=lane, K {0..7,16..23}; lanes 16-31 K {8..15,24..31}
//    D: vgpr r -> row r (lanes 0-15) / r+8 (lanes 16-31)
// ---------------------------------------------------------------------------
template <int HC>
__global__ __launch_bounds__(128)
void gat_gemm_wmma(const float* __restrict__ x,
                   const float* __restrict__ Wl,
                   const float* __restrict__ Wr,
                   float* __restrict__ xl,
                   float* __restrict__ xr) {
    constexpr int TPW = HC / 32;            // 16x16 n-tiles per wave (4 or 2)
    __shared__ __align__(16) float    sxraw[16 * K_DIM];       // 8 KB
    __shared__ __align__(16) _Float16 sxh[16][136];            // 4.25 KB
    __shared__ __align__(16) _Float16 sWl[HC][136];            // <= 34 KB
    __shared__ __align__(16) _Float16 sWr[HC][136];            // <= 34 KB

    const int tid    = threadIdx.x;
    const int lane   = tid & 31;
    const int wave   = tid >> 5;
    const int tile_m = blockIdx.x;

    // ---- stage x tile: async global->LDS (raw fp32), 512 x 16B chunks ----
    {
        const unsigned lds_base =
            (unsigned)(size_t)(void*)&sxraw[0];
        const unsigned long long gbase =
            (unsigned long long)(const void*)(x + (size_t)tile_m * 16 * K_DIM);
#pragma unroll
        for (int c = tid; c < 512; c += 128) {
            unsigned           laddr = lds_base + (unsigned)c * 16u;
            unsigned long long gaddr = gbase + (unsigned long long)c * 16ull;
            asm volatile("global_load_async_to_lds_b128 %0, %1, off"
                         :: "v"(laddr), "v"(gaddr) : "memory");
        }
        asm volatile("s_wait_asynccnt 0" ::: "memory");
    }
    // ---- stage W (both sides), transposed + f16 ----
    for (int idx = tid; idx < K_DIM * HC; idx += 128) {
        const int k = idx / HC;             // consecutive tid -> same k row
        const int n = idx % HC;             // coalesced global reads
        sWl[n][k] = (_Float16)Wl[(size_t)k * HC + n];
        sWr[n][k] = (_Float16)Wr[(size_t)k * HC + n];
    }
    __syncthreads();
    // ---- convert x tile fp32 -> f16 (padded rows) ----
    for (int i = tid; i < 16 * K_DIM; i += 128) {
        sxh[i >> 7][i & 127] = (_Float16)sxraw[i];
    }
    __syncthreads();

    const int side  = wave >> 1;            // 0: Wl->xl, 1: Wr->xr
    const int nhalf = wave & 1;             // which half of the columns
    const _Float16(*sW)[136] = side ? sWr : sWl;
    float* out = side ? xr : xl;

    const int half = lane >> 4;
    const int l15  = lane & 15;
    const int akb  = half * 8;              // A k-base
    const int bkb  = half * 8;              // B k-base

    v8f acc[TPW];
#pragma unroll
    for (int t = 0; t < TPW; ++t) acc[t] = (v8f){};

    const _Float16* arow = &sxh[l15][0];
#pragma unroll
    for (int k0 = 0; k0 < K_DIM; k0 += 32) {
        const v8h alo = *(const v8h*)(arow + k0 + akb);
        const v8h ahi = *(const v8h*)(arow + k0 + akb + 16);
        v16h af;
#pragma unroll
        for (int j = 0; j < 8; ++j) { af[j] = alo[j]; af[j + 8] = ahi[j]; }
#pragma unroll
        for (int t = 0; t < TPW; ++t) {
            const int n = (nhalf * TPW + t) * 16 + l15;
            const _Float16* brow = &sW[n][0];
            const v8h blo = *(const v8h*)(brow + k0 + bkb);
            const v8h bhi = *(const v8h*)(brow + k0 + bkb + 16);
            v16h bf;
#pragma unroll
            for (int j = 0; j < 8; ++j) { bf[j] = blo[j]; bf[j + 8] = bhi[j]; }
            acc[t] = __builtin_amdgcn_wmma_f32_16x16x32_f16(
                false, af, false, bf, (short)0, acc[t], false, false);
        }
    }

#pragma unroll
    for (int t = 0; t < TPW; ++t) {
        const int n = (nhalf * TPW + t) * 16 + l15;
#pragma unroll
        for (int r = 0; r < 8; ++r) {
            const int mm = tile_m * 16 + r + half * 8;
            out[(size_t)mm * HC + n] = acc[t][r];
        }
    }
}

// ---------------------------------------------------------------------------
// Per-layer init: maxbuf = -1e30, den = 0, agg = 0
// ---------------------------------------------------------------------------
__global__ void gat_init(float* __restrict__ maxb, float* __restrict__ den,
                         float* __restrict__ agg, int nmd, int nagg) {
    const int i = blockIdx.x * blockDim.x + threadIdx.x;
    if (i < nmd) { maxb[i] = -1e30f; den[i] = 0.0f; }
    if (i < nagg) agg[i] = 0.0f;
}

// ---------------------------------------------------------------------------
// score[e,h] = att_h . leaky_relu(xl[src]+xr[dst]) fused with segment max.
// One wave per edge; each lane owns a contiguous 2H-float chunk (b128/b64
// loads), reduction within 32/H-lane head groups.
// ---------------------------------------------------------------------------
template <int H>
__global__ void gat_edge_score(const float* __restrict__ xl,
                               const float* __restrict__ xr,
                               const int* __restrict__ srcs,
                               const int* __restrict__ dsts,
                               const float* __restrict__ att,
                               float* __restrict__ score,
                               float* __restrict__ maxb) {
    const int e = blockIdx.x * (blockDim.x >> 5) + (threadIdx.x >> 5);
    if (e >= E_EDGES) return;
    const int lane = threadIdx.x & 31;
    constexpr int HC  = H * HIDC;
    constexpr int EPL = 2 * H;      // floats per lane
    constexpr int G   = 32 / H;     // lanes per head group
    const int s = srcs[e], d = dsts[e];
    const float* pl = xl + (size_t)s * HC + lane * EPL;
    const float* pr = xr + (size_t)d * HC + lane * EPL;
    const float* pa = att + lane * EPL;

    float p = 0.0f;
#pragma unroll
    for (int j = 0; j < EPL; ++j) {
        float v = pl[j] + pr[j];
        v = (v > 0.0f) ? v : 0.2f * v;        // leaky_relu(0.2)
        p += v * pa[j];
    }
#pragma unroll
    for (int off = 1; off < G; off <<= 1) p += __shfl_xor(p, off, 32);
    if ((lane & (G - 1)) == 0) {
        const int h = lane / G;
        score[(size_t)e * H + h] = p;
        atomicMaxF(&maxb[(size_t)d * H + h], p);
    }
}

// ---------------------------------------------------------------------------
// ex = exp(score - max[dst]); den[dst] += ex   (overwrites score with ex)
// ---------------------------------------------------------------------------
__global__ void gat_edge_exp(const int* __restrict__ dsts,
                             float* __restrict__ score,
                             const float* __restrict__ maxb,
                             float* __restrict__ den, int H) {
    const int t = blockIdx.x * blockDim.x + threadIdx.x;
    if (t >= E_EDGES * H) return;
    const int e = t / H, h = t - e * H;
    const int d = dsts[e];
    const float ex = __expf(score[t] - maxb[(size_t)d * H + h]);
    score[t] = ex;
    atomicAdd(&den[(size_t)d * H + h], ex);
}

// ---------------------------------------------------------------------------
// agg[dst] += xl[src] * (ex / (den[dst]+eps)); wave/edge, lane-contiguous
// chunks -> coalesced global_atomic_add_f32 stream.
// ---------------------------------------------------------------------------
template <int H>
__global__ void gat_edge_agg(const float* __restrict__ xl,
                             const float* __restrict__ ex,
                             const float* __restrict__ den,
                             const int* __restrict__ srcs,
                             const int* __restrict__ dsts,
                             float* __restrict__ agg) {
    const int e = blockIdx.x * (blockDim.x >> 5) + (threadIdx.x >> 5);
    if (e >= E_EDGES) return;
    const int lane = threadIdx.x & 31;
    constexpr int HC  = H * HIDC;
    constexpr int EPL = 2 * H;
    const int s = srcs[e], d = dsts[e];
    const int h = (lane * EPL) >> 6;
    const float alpha = ex[(size_t)e * H + h] /
                        (den[(size_t)d * H + h] + 1e-16f);
    const float* pl = xl  + (size_t)s * HC + lane * EPL;
    float*       pd = agg + (size_t)d * HC + lane * EPL;
#pragma unroll
    for (int j = 0; j < EPL; ++j) atomicAdd(&pd[j], pl[j] * alpha);
}

// ---------------------------------------------------------------------------
// In-place: agg = (relu ? max(agg + b, 0) : agg + b)
// ---------------------------------------------------------------------------
__global__ void gat_bias_act(float* __restrict__ agg,
                             const float* __restrict__ b, int HC, int do_relu) {
    const int t = blockIdx.x * blockDim.x + threadIdx.x;
    if (t >= N_NODES * HC) return;
    float v = agg[t] + b[t % HC];
    agg[t] = do_relu ? fmaxf(v, 0.0f) : v;
}

// ---------------------------------------------------------------------------
// host side
// ---------------------------------------------------------------------------
template <int H>
static void run_layer(const float* x, const float* Wl, const float* Wr,
                      const float* att, const float* b, float* out,
                      int do_relu, float* xl, float* xr, float* score,
                      float* maxb, float* den, const int* srcs,
                      const int* dsts, hipStream_t stream) {
    constexpr int HC = H * HIDC;
    gat_gemm_wmma<HC><<<dim3(N_NODES / 16, 1, 1), 128, 0, stream>>>(
        x, Wl, Wr, xl, xr);
    const int nagg = N_NODES * HC;
    gat_init<<<(nagg + 255) / 256, 256, 0, stream>>>(maxb, den, out,
                                                     N_NODES * H, nagg);
    gat_edge_score<H><<<(E_EDGES + 7) / 8, 256, 0, stream>>>(
        xl, xr, srcs, dsts, att, score, maxb);
    gat_edge_exp<<<(E_EDGES * H + 255) / 256, 256, 0, stream>>>(
        dsts, score, maxb, den, H);
    gat_edge_agg<H><<<(E_EDGES + 7) / 8, 256, 0, stream>>>(
        xl, score, den, srcs, dsts, out);
    gat_bias_act<<<(nagg + 255) / 256, 256, 0, stream>>>(out, b, HC, do_relu);
}

extern "C" void kernel_launch(void* const* d_in, const int* in_sizes, int n_in,
                              void* d_out, int out_size, void* d_ws,
                              size_t ws_size, hipStream_t stream) {
    (void)in_sizes; (void)n_in; (void)out_size; (void)ws_size;
    const float* x    = (const float*)d_in[0];
    const int*   ei   = (const int*)d_in[1];
    const int*   srcs = ei;                 // edge_index[0]
    const int*   dsts = ei + E_EDGES;       // edge_index[1]

    const float* Wl[4]  = {(const float*)d_in[2],  (const float*)d_in[6],
                           (const float*)d_in[10], (const float*)d_in[14]};
    const float* Wr[4]  = {(const float*)d_in[3],  (const float*)d_in[7],
                           (const float*)d_in[11], (const float*)d_in[15]};
    const float* att[4] = {(const float*)d_in[4],  (const float*)d_in[8],
                           (const float*)d_in[12], (const float*)d_in[16]};
    const float* bia[4] = {(const float*)d_in[5],  (const float*)d_in[9],
                           (const float*)d_in[13], (const float*)d_in[17]};

    char*  ws  = (char*)d_ws;
    size_t off = 0;
    auto carve = [&](size_t elems) {
        float* p = (float*)(ws + off);
        off = (off + elems * sizeof(float) + 255) & ~(size_t)255;
        return p;
    };
    float* xl    = carve((size_t)N_NODES * 128);
    float* xr    = carve((size_t)N_NODES * 128);
    float* bufA  = carve((size_t)N_NODES * 128);
    float* bufB  = carve((size_t)N_NODES * 128);
    float* score = carve((size_t)E_EDGES * 2);
    float* maxb  = carve((size_t)N_NODES * 2);
    float* den   = carve((size_t)N_NODES * 2);

    float* outf = (float*)d_out;

    run_layer<2>(x,    Wl[0], Wr[0], att[0], bia[0], bufA, 1,
                 xl, xr, score, maxb, den, srcs, dsts, stream);
    run_layer<2>(bufA, Wl[1], Wr[1], att[1], bia[1], bufB, 1,
                 xl, xr, score, maxb, den, srcs, dsts, stream);
    run_layer<2>(bufB, Wl[2], Wr[2], att[2], bia[2], bufA, 1,
                 xl, xr, score, maxb, den, srcs, dsts, stream);
    run_layer<1>(bufA, Wl[3], Wr[3], att[3], bia[3], outf, 0,
                 xl, xr, score, maxb, den, srcs, dsts, stream);
}
